// Model_Relation_Self_Attention_36679020708364
// MI455X (gfx1250) — matrix operand order repaired and checked by reading, not compiled, not run on hardware
//
#include <hip/hip_runtime.h>
#include <hip/hip_bf16.h>
#include <stdint.h>

// ---------------------------------------------------------------------------
// Types for CDNA5 WMMA (wave32) + TDM descriptors
// ---------------------------------------------------------------------------
typedef __bf16 bf16;
typedef __attribute__((ext_vector_type(16))) bf16  v16bf;
typedef __attribute__((ext_vector_type(8)))  bf16  v8bf;
typedef __attribute__((ext_vector_type(8)))  float v8f;
typedef unsigned int u32v4 __attribute__((ext_vector_type(4)));
typedef int          i32v8 __attribute__((ext_vector_type(8)));
typedef int          i32v4 __attribute__((ext_vector_type(4)));

static constexpr int Bc = 8, Sc = 512, Ec = 1024, Hc = 16, HDc = 64, FFc = 4096, Pc = 64;

#if defined(__gfx1250__) && __has_builtin(__builtin_amdgcn_tensor_load_to_lds) && \
    __has_builtin(__builtin_amdgcn_s_wait_tensorcnt)
#define USE_TDM 1
#else
#define USE_TDM 0
#endif

__device__ __forceinline__ bf16 f2bf(float f) {
  unsigned u = __float_as_uint(f);
  u += 0x7FFFu + ((u >> 16) & 1u);      // round-to-nearest-even
  unsigned short hs = (unsigned short)(u >> 16);
  return __builtin_bit_cast(bf16, hs);
}
__device__ __forceinline__ float bf2f(bf16 h) {
  unsigned short hs = __builtin_bit_cast(unsigned short, h);
  return __uint_as_float(((unsigned)hs) << 16);
}

// A-fragment (16 rows x K-chunk 32, 16-bit): lane<16 row=lane K=0..7,16..23;
// lane>=16 row=lane-16 K=8..15,24..31.
__device__ __forceinline__ v16bf ld_afrag(const bf16* Abase, int lda, int lane) {
  int lr = lane & 15, hf = lane >> 4;
  const bf16* p = Abase + lr * lda + hf * 8;
  v8bf lo = *(const v8bf*)p;
  v8bf hi = *(const v8bf*)(p + 16);
  v16bf a;
#pragma unroll
  for (int i = 0; i < 8; ++i) { a[i] = lo[i]; a[i + 8] = hi[i]; }
  return a;
}
// B-fragment (32x16, 16-bit): lane<16 col=lane K=0..7,16..23;
// lane>=16 col=lane-16 K=8..15,24..31.
__device__ __forceinline__ v16bf ld_bfrag(const bf16* Bbase, int ldb, int lane) {
  int lr = lane & 15, hf = lane >> 4;
  const bf16* p = Bbase + lr;
  v16bf b;
#pragma unroll
  for (int i = 0; i < 8; ++i) { b[i] = p[(hf * 8 + i) * ldb]; b[i + 8] = p[(16 + hf * 8 + i) * ldb]; }
  return b;
}

#if USE_TDM
// Issue a 2-D tile load (bf16, rows packed contiguously into LDS) via the
// Tensor Data Mover. D# layout per CDNA5 ISA ch.8 (group0 128b, group1 256b,
// trailing groups zero => 2-D tensor). Tracked by TENSORcnt.
// This toolchain's builtin takes 6 args (g0, g1, g2, g3, g4, cpol).
__device__ __forceinline__ void tdm_load_2d(unsigned lds_off, const bf16* gptr,
                                            unsigned tile_w, unsigned tile_h,
                                            unsigned tens_w, unsigned tens_h,
                                            unsigned stride_elems) {
  uint64_t ga = (uint64_t)(uintptr_t)gptr;
  u32v4 g0;
  g0[0] = 1u;                                            // count=1, user mode
  g0[1] = lds_off;                                       // lds_addr (bytes)
  g0[2] = (unsigned)ga;                                  // global_addr[31:0]
  g0[3] = (unsigned)((ga >> 32) & 0x1FFFFFFu) | (2u << 30); // addr[56:32] | type=2
  i32v8 g1;
  g1[0] = (int)(1u << 16);                               // data_size=1 (2 bytes)
  g1[1] = (int)((tens_w & 0xFFFFu) << 16);               // tensor_dim0[15:0]
  g1[2] = (int)((tens_w >> 16) | ((tens_h & 0xFFFFu) << 16)); // dim0 hi / dim1 lo
  g1[3] = (int)((tens_h >> 16) | (tile_w << 16));        // dim1 hi / tile_dim0
  g1[4] = (int)(tile_h & 0xFFFFu);                       // tile_dim1 (tile_dim2=0)
  g1[5] = (int)stride_elems;                             // tensor_dim0_stride lo32
  g1[6] = 0;
  g1[7] = 0;
  i32v4 z4 = {0, 0, 0, 0};
  i32v8 z8 = {0, 0, 0, 0, 0, 0, 0, 0};
  __builtin_amdgcn_tensor_load_to_lds(g0, g1, z4, z4, z8, 0);
}
#endif

// ---------------------------------------------------------------------------
// WMMA GEMM, bf16 operands in global memory:
//   C = act(A[M,K] @ W[K,N] + bias [+ res]);  f32 and/or bf16 outputs.
// 128 threads / 4 waves, 64x64 tile, K chunks of 32, TDM double-buffered.
// ---------------------------------------------------------------------------
__global__ __launch_bounds__(128)
void gemm_bf16_kernel(const bf16* __restrict__ A, const bf16* __restrict__ W,
                      const float* __restrict__ bias, const float* __restrict__ res,
                      float* __restrict__ Cf, bf16* __restrict__ Cbf,
                      int M, int N, int K, int relu)
{
  __shared__ __align__(16) bf16 As[2][64 * 32];
  __shared__ __align__(16) bf16 Bs[2][32 * 64];
  const int tid = threadIdx.x, wave = tid >> 5, lane = tid & 31;
  const int lr = lane & 15, half = lane >> 4;
  const int n0 = blockIdx.x * 64, m0 = blockIdx.y * 64;
  const int nk = K >> 5;

  v8f zero = {0.f,0.f,0.f,0.f,0.f,0.f,0.f,0.f};
  v8f acc[4];
#pragma unroll
  for (int t = 0; t < 4; ++t) acc[t] = zero;

#if USE_TDM
  if (wave == 0) {
    tdm_load_2d((unsigned)(uintptr_t)&As[0][0], A + (size_t)m0 * K, 32, 64, K, M, K);
    tdm_load_2d((unsigned)(uintptr_t)&Bs[0][0], W + n0,             64, 32, N, K, N);
  }
#endif
  for (int kc = 0; kc < nk; ++kc) {
    const int cur = kc & 1;
#if USE_TDM
    if (wave == 0) {
      if (kc + 1 < nk) {
        const int nxt = cur ^ 1;
        tdm_load_2d((unsigned)(uintptr_t)&As[nxt][0],
                    A + (size_t)m0 * K + (size_t)(kc + 1) * 32, 32, 64, K, M, K);
        tdm_load_2d((unsigned)(uintptr_t)&Bs[nxt][0],
                    W + (size_t)(kc + 1) * 32 * N + n0,         64, 32, N, K, N);
        __builtin_amdgcn_s_wait_tensorcnt(2);   // chunk kc landed (in-order)
      } else {
        __builtin_amdgcn_s_wait_tensorcnt(0);
      }
    }
    __syncthreads();
#else
#pragma unroll
    for (int e = 0; e < 16; ++e) {
      int idx = e * 128 + tid;
      As[cur][idx] = A[(size_t)(m0 + (idx >> 5)) * K + kc * 32 + (idx & 31)];
      Bs[cur][idx] = W[(size_t)(kc * 32 + (idx >> 6)) * N + n0 + (idx & 63)];
    }
    __syncthreads();
#endif
    v16bf a = ld_afrag(&As[cur][0] + wave * 16 * 32, 32, lane);
#pragma unroll
    for (int t = 0; t < 4; ++t) {
      v16bf b = ld_bfrag(&Bs[cur][0] + t * 16, 64, lane);
      acc[t] = __builtin_amdgcn_wmma_f32_16x16x32_bf16(false, a, false, b,
                                                       (short)0, acc[t], false, false);
    }
    __syncthreads();
  }

#pragma unroll
  for (int t = 0; t < 4; ++t) {
    int col = n0 + t * 16 + lr;
    float bv = bias[col];
#pragma unroll
    for (int r = 0; r < 8; ++r) {
      int row = m0 + wave * 16 + r + half * 8;
      float v = acc[t][r] + bv;
      if (res) v += res[(size_t)row * N + col];
      if (relu) v = fmaxf(v, 0.f);
      if (Cf)  Cf [(size_t)row * N + col] = v;
      if (Cbf) Cbf[(size_t)row * N + col] = f2bf(v);
    }
  }
}

// ---------------------------------------------------------------------------
// f32 -> bf16 conversion (weights / x), grid-stride.
// ---------------------------------------------------------------------------
__global__ __launch_bounds__(256)
void cvt_bf16_kernel(const float* __restrict__ in, bf16* __restrict__ out, int n) {
  for (int i = blockIdx.x * 256 + threadIdx.x; i < n; i += gridDim.x * 256)
    out[i] = f2bf(in[i]);
}

// ---------------------------------------------------------------------------
// Fused relative-position attention for one (b, h, 64-row i-tile).
// R never materialized: q.R collapses to qp[i, idx], aw.R to hist @ pos_emb.
// LDS (dynamic, 250880 B): scores f32 [64][512] | aw bf16 [64][512] |
//   q bf16 [64][72] | k^T/v bf16 [64][72] | qp/hist f32 [64][132] | dt i32 [512]
// ---------------------------------------------------------------------------
__global__ __launch_bounds__(128)
void attention_kernel(const float* __restrict__ q, const float* __restrict__ k,
                      const float* __restrict__ v, const float* __restrict__ pos_emb,
                      const int* __restrict__ dt, const float* __restrict__ dist,
                      bf16* __restrict__ ctxbf)
{
  extern __shared__ __align__(16) char smem[];
  float* sc   = (float*)(smem);
  bf16*  aw   = (bf16*)(smem + 131072);
  bf16*  qs   = (bf16*)(smem + 196608);
  bf16*  kvs  = (bf16*)(smem + 205824);
  float* hist = (float*)(smem + 215040);
  int*   dts  = (int*)(smem + 248832);

  const int i0 = blockIdx.x * 64;
  const int h  = blockIdx.y;
  const int b  = blockIdx.z;
  const int tid = threadIdx.x, wave = tid >> 5, lane = tid & 31;
  const int lr = lane & 15, half = lane >> 4;

  for (int j = tid; j < Sc; j += 128) dts[j] = dt[b * Sc + j];

  const float* qg = q + ((size_t)(b * Sc + i0)) * Ec + h * HDc;
#pragma unroll
  for (int e = 0; e < 32; ++e) {
    int idx = e * 128 + tid;
    int r = idx >> 6, d = idx & 63;
    qs[r * 72 + d] = f2bf(qg[(size_t)r * Ec + d]);
  }
  __syncthreads();

  for (int idx = tid; idx < 64 * 129; idx += 128) {   // qp[i][c] = q_i . pos_emb[c]
    int i = idx / 129, c = idx % 129;
    const float* pe = pos_emb + c * HDc;
    float s = 0.f;
    for (int d = 0; d < HDc; ++d) s += bf2f(qs[i * 72 + d]) * pe[d];
    hist[i * 132 + c] = s;
  }
  __syncthreads();

  v8f zero = {0.f,0.f,0.f,0.f,0.f,0.f,0.f,0.f};
  v8f acc[4];

  for (int jc = 0; jc < 8; ++jc) {                    // scores
    const int j0 = jc * 64;
    const float* kg = k + ((size_t)(b * Sc + j0)) * Ec + h * HDc;
#pragma unroll
    for (int e = 0; e < 32; ++e) {                    // k staged transposed
      int idx = e * 128 + tid;
      int j = idx >> 6, d = idx & 63;
      kvs[d * 72 + j] = f2bf(kg[(size_t)j * Ec + d]);
    }
    __syncthreads();
#pragma unroll
    for (int t = 0; t < 4; ++t) acc[t] = zero;
#pragma unroll
    for (int kc = 0; kc < 2; ++kc) {
      v16bf a = ld_afrag(qs + wave * 16 * 72 + kc * 32, 72, lane);
#pragma unroll
      for (int t = 0; t < 4; ++t) {
        v16bf bb = ld_bfrag(kvs + kc * 32 * 72 + t * 16, 72, lane);
        acc[t] = __builtin_amdgcn_wmma_f32_16x16x32_bf16(false, a, false, bb,
                                                         (short)0, acc[t], false, false);
      }
    }
#pragma unroll
    for (int t = 0; t < 4; ++t) {
      int jabs = j0 + t * 16 + lr;
      int dtj = dts[jabs];
#pragma unroll
      for (int r = 0; r < 8; ++r) {
        int il = wave * 16 + r + half * 8;
        int rel = dtj - dts[i0 + il];
        rel = rel < -Pc ? -Pc : (rel > Pc ? Pc : rel);
        float s = acc[t][r] + hist[il * 132 + rel + Pc];
        s = s * 0.125f + 0.6f * dist[((size_t)b * Sc + i0 + il) * Sc + jabs];
        sc[il * 512 + jabs] = s;
      }
    }
    __syncthreads();
  }

  for (int idx = tid; idx < 64 * 132; idx += 128) hist[idx] = 0.f;
  __syncthreads();
  {                                                   // softmax + idx histogram
    const int row = wave * 16 + lr;
    float* srow = sc + row * 512;
    const int jb = half * 256;
    float m = -3.4e38f;
    for (int j = 0; j < 256; ++j) m = fmaxf(m, srow[jb + j]);
    m = fmaxf(m, __shfl_xor(m, 16, 32));
    float sum = 0.f;
    for (int j = 0; j < 256; ++j) {
      float e = __expf(srow[jb + j] - m);
      srow[jb + j] = e;
      sum += e;
    }
    sum += __shfl_xor(sum, 16, 32);
    float rinv = 1.f / sum;
    int dti = dts[i0 + row];
    bf16* arow = aw + row * 512;
    for (int j = 0; j < 256; ++j) {
      int jabs = jb + j;
      float aval = srow[jabs] * rinv;
      arow[jabs] = f2bf(aval);
      int rel = dts[jabs] - dti;
      rel = rel < -Pc ? -Pc : (rel > Pc ? Pc : rel);
      atomicAdd(&hist[row * 132 + rel + Pc], aval);   // ds_add_f32
    }
  }
  __syncthreads();

#pragma unroll
  for (int t = 0; t < 4; ++t) acc[t] = zero;
  for (int jc = 0; jc < 16; ++jc) {                   // attn = aw @ v
    const int j0 = jc * 32;
    const float* vg = v + ((size_t)(b * Sc + j0)) * Ec + h * HDc;
#pragma unroll
    for (int e = 0; e < 16; ++e) {
      int idx = e * 128 + tid;
      int j = idx >> 6, d = idx & 63;
      kvs[j * 72 + d] = f2bf(vg[(size_t)j * Ec + d]);
    }
    __syncthreads();
    v16bf a = ld_afrag(aw + wave * 16 * 512 + j0, 512, lane);
#pragma unroll
    for (int t = 0; t < 4; ++t) {
      v16bf bb = ld_bfrag(kvs + t * 16, 72, lane);
      acc[t] = __builtin_amdgcn_wmma_f32_16x16x32_bf16(false, a, false, bb,
                                                       (short)0, acc[t], false, false);
    }
    __syncthreads();
  }

#pragma unroll
  for (int t = 0; t < 4; ++t) {                       // + hist @ pos_emb, bf16 out
    int d = t * 16 + lr;
#pragma unroll
    for (int r = 0; r < 8; ++r) {
      int il = wave * 16 + r + half * 8;
      float s = acc[t][r];
      const float* hr = hist + il * 132;
      for (int c = 0; c < 129; ++c) s += hr[c] * pos_emb[c * HDc + d];
      ctxbf[((size_t)b * Sc + i0 + il) * Ec + h * HDc + d] = f2bf(s);
    }
  }
}

// ---------------------------------------------------------------------------
// LayerNorm (last dim 1024); optional bf16 copy for the next GEMM's A operand.
// ---------------------------------------------------------------------------
__global__ __launch_bounds__(256)
void layernorm_kernel(const float* __restrict__ x, const float* __restrict__ g,
                      const float* __restrict__ bta, float* __restrict__ out,
                      bf16* __restrict__ out_bf)
{
  __shared__ float red[256];
  const int row = blockIdx.x;
  const float* xr = x + (size_t)row * Ec;
  float s = 0.f;
  for (int i = threadIdx.x; i < Ec; i += 256) s += xr[i];
  red[threadIdx.x] = s; __syncthreads();
  for (int o = 128; o > 0; o >>= 1) { if (threadIdx.x < o) red[threadIdx.x] += red[threadIdx.x + o]; __syncthreads(); }
  float mu = red[0] * (1.f / Ec);
  __syncthreads();
  float vv = 0.f;
  for (int i = threadIdx.x; i < Ec; i += 256) { float d = xr[i] - mu; vv += d * d; }
  red[threadIdx.x] = vv; __syncthreads();
  for (int o = 128; o > 0; o >>= 1) { if (threadIdx.x < o) red[threadIdx.x] += red[threadIdx.x + o]; __syncthreads(); }
  float rs = rsqrtf(red[0] * (1.f / Ec) + 1e-5f);
  for (int i = threadIdx.x; i < Ec; i += 256) {
    float val = (xr[i] - mu) * rs * g[i] + bta[i];
    out[(size_t)row * Ec + i] = val;
    if (out_bf) out_bf[(size_t)row * Ec + i] = f2bf(val);
  }
}

// ---------------------------------------------------------------------------
extern "C" void kernel_launch(void* const* d_in, const int* in_sizes, int n_in,
                              void* d_out, int out_size, void* d_ws, size_t ws_size,
                              hipStream_t stream) {
  (void)in_sizes; (void)n_in; (void)out_size; (void)ws_size;
  const float* x      = (const float*)d_in[0];
  const float* dist   = (const float*)d_in[1];
  const int*   dts    = (const int*)  d_in[2];
  const float* pos    = (const float*)d_in[3];
  const float* wq_w   = (const float*)d_in[4];  const float* wq_b = (const float*)d_in[5];
  const float* wk_w   = (const float*)d_in[6];  const float* wk_b = (const float*)d_in[7];
  const float* wv_w   = (const float*)d_in[8];  const float* wv_b = (const float*)d_in[9];
  const float* wo_w   = (const float*)d_in[10]; const float* wo_b = (const float*)d_in[11];
  const float* ff1_w  = (const float*)d_in[12]; const float* ff1_b = (const float*)d_in[13];
  const float* ff2_w  = (const float*)d_in[14]; const float* ff2_b = (const float*)d_in[15];
  const float* ln1_g  = (const float*)d_in[16]; const float* ln1_b = (const float*)d_in[17];
  const float* ln2_g  = (const float*)d_in[18]; const float* ln2_b = (const float*)d_in[19];

  const size_t MB = 1024u * 1024u;
  char* ws = (char*)d_ws;
  // f32 activations
  float* q    = (float*)(ws + 0 * MB);     // 16 MB (dead after attention)
  float* kk   = (float*)(ws + 16 * MB);    // 16 MB (dead after attention)
  float* vv   = (float*)(ws + 32 * MB);    // 16 MB (dead after attention)
  bf16*  midb = (bf16*) (ws + 0 * MB);     // 32 MB, reuses q/k
  bf16*  h_bf = (bf16*) (ws + 32 * MB);    //  8 MB, reuses v
  bf16*  ctxb = (bf16*) (ws + 48 * MB);    //  8 MB
  bf16*  x_bf = (bf16*) (ws + 56 * MB);    //  8 MB
  // bf16 weights (converted once per launch)
  bf16* wq_bf  = (bf16*)(ws + 64 * MB);    // 2 MB
  bf16* wk_bf  = (bf16*)(ws + 66 * MB);
  bf16* wv_bf  = (bf16*)(ws + 68 * MB);
  bf16* wo_bf  = (bf16*)(ws + 70 * MB);
  bf16* ff1_bf = (bf16*)(ws + 72 * MB);    // 8 MB
  bf16* ff2_bf = (bf16*)(ws + 80 * MB);    // 8 MB
  float* y1    = (float*)(ws + 88 * MB);   // 16 MB (y2 reuses after LN1)
  float* y2    = (float*)(ws + 88 * MB);
  float* hbuf  = (float*)(ws + 104 * MB);  // 16 MB   (peak ws: 120 MB)

  const int M  = Bc * Sc;                  // 4096
  const int EE = Ec * Ec, EF = Ec * FFc;
  dim3 blk(128);
  dim3 gE(Ec / 64, M / 64);
  dim3 gFF(FFc / 64, M / 64);

  // One-shot bf16 conversions (operands for the TDM-fed GEMMs)
  cvt_bf16_kernel<<<2048, 256, 0, stream>>>(x,     x_bf,   M * Ec);
  cvt_bf16_kernel<<<2048, 256, 0, stream>>>(wq_w,  wq_bf,  EE);
  cvt_bf16_kernel<<<2048, 256, 0, stream>>>(wk_w,  wk_bf,  EE);
  cvt_bf16_kernel<<<2048, 256, 0, stream>>>(wv_w,  wv_bf,  EE);
  cvt_bf16_kernel<<<2048, 256, 0, stream>>>(wo_w,  wo_bf,  EE);
  cvt_bf16_kernel<<<2048, 256, 0, stream>>>(ff1_w, ff1_bf, EF);
  cvt_bf16_kernel<<<2048, 256, 0, stream>>>(ff2_w, ff2_bf, EF);

  // QKV projections (TDM-staged WMMA)
  gemm_bf16_kernel<<<gE, blk, 0, stream>>>(x_bf, wq_bf, wq_b, nullptr, q,  nullptr, M, Ec, Ec, 0);
  gemm_bf16_kernel<<<gE, blk, 0, stream>>>(x_bf, wk_bf, wk_b, nullptr, kk, nullptr, M, Ec, Ec, 0);
  gemm_bf16_kernel<<<gE, blk, 0, stream>>>(x_bf, wv_bf, wv_b, nullptr, vv, nullptr, M, Ec, Ec, 0);

  // Fused relative attention -> ctx (bf16, [B,S,E] concat layout)
  dim3 ga(Sc / 64, Hc, Bc);
  constexpr size_t ATT_SMEM = 250880;
  attention_kernel<<<ga, blk, ATT_SMEM, stream>>>(q, kk, vv, pos, dts, dist, ctxb);

  // Output projection + residual(x), LN1 (emits f32 h and bf16 h for ff1)
  gemm_bf16_kernel<<<gE, blk, 0, stream>>>(ctxb, wo_bf, wo_b, x, y1, nullptr, M, Ec, Ec, 0);
  layernorm_kernel<<<M, 256, 0, stream>>>(y1, ln1_g, ln1_b, hbuf, h_bf);

  // FFN: ff1 (ReLU, bf16-only output) -> ff2 (+residual h) -> LN2
  gemm_bf16_kernel<<<gFF, blk, 0, stream>>>(h_bf, ff1_bf, ff1_b, nullptr, nullptr, midb, M, FFc, Ec, 1);
  gemm_bf16_kernel<<<gE,  blk, 0, stream>>>(midb, ff2_bf, ff2_b, hbuf, y2, nullptr, M, Ec, FFc, 0);
  layernorm_kernel<<<M, 256, 0, stream>>>(y2, ln2_g, ln2_b, (float*)d_out, nullptr);
}
